// MSABlock_11373073400388
// MI455X (gfx1250) — matrix operand order repaired and checked by reading, not compiled, not run on hardware
//
#include <hip/hip_runtime.h>

typedef unsigned short u16;
typedef __attribute__((ext_vector_type(16))) __bf16 v16bf;
typedef __attribute__((ext_vector_type(8)))  float  v8f;

#define BM 128
#define BN 64
#define BK 32

// ---------------- device helpers ----------------
static __device__ __forceinline__ u16 f2bf(float f) {
  unsigned int u = __float_as_uint(f);
  u += 0x7fffu + ((u >> 16) & 1u);            // round-to-nearest-even
  return (u16)(u >> 16);
}

struct U32x8 { uint4 a, b; };
static __device__ __forceinline__ v16bf mk16(uint4 x, uint4 y) {
  U32x8 t; t.a = x; t.b = y;
  return __builtin_bit_cast(v16bf, t);
}

// gfx1250 async global->LDS copy (ASYNCcnt-tracked). vdst = 32-bit LDS byte
// address (low 32 bits of the generic shared pointer), vaddr = 64-bit global.
static __device__ __forceinline__ void async_ld_b128(void* lds, const void* gp) {
  unsigned l = (unsigned)(unsigned long long)lds;
  asm volatile("global_load_async_to_lds_b128 %0, %1, off"
               :: "v"(l), "v"(gp) : "memory");
}
#define WAIT_ASYNC(n) asm volatile("s_wait_asynccnt " #n ::: "memory")

// ---------------- generic batched WMMA GEMM ----------------
// C[M,N] (+)= act(A[M,K] * B[K,N] + bias) * rowscale * emul   (all optional)
// A bf16 row-major [M,K] (M must be covered by whole 128-row blocks; all call
// sites use M % 128 == 0 or M=256). B: bNK=1 -> stored [N,K] row-major (fast
// async path, row index clamped; cols >= N are masked in the epilogue),
// bNK=0 -> stored [K,N] row-major (strided sync path, attention A*V only).
// Batch offset: b_out=bo/H, b_in=bo%H -> ptr + b_out*s?o + b_in*s?i.
// cmode=1: C index = (m*ldc+n)*sCi + bo   (channel-last pack, tri-mul einsum)
// dmode=1: D index = ((m>>5)*256+(n>>5))*1024 + (m&31)*32 + (n&31)  (OPM pack)
__global__ __launch_bounds__(256) void k_gemm(
    const u16* __restrict__ A, int lda, long long sAo, long long sAi,
    const u16* __restrict__ B, int ldb, long long sBo, long long sBi, int bNK,
    float* C, int ldc, long long sCo, long long sCi, int cmode,
    u16* D, int ldd, long long sDo, long long sDi, int dmode,
    const float* __restrict__ bias, const float* __restrict__ rowscale,
    const float* __restrict__ emul,
    int M, int N, int K, int H, int act, int accum)
{
  __shared__ __align__(16) u16 As[2][BM][BK];   // 16 KB ping-pong
  __shared__ __align__(16) u16 Bs[2][BN][BK];   //  8 KB ping-pong
  const int bo    = blockIdx.z;
  const int b_out = bo / H;
  const int b_in  = bo - b_out * H;
  const u16* Ab = A + (long long)b_out * sAo + (long long)b_in * sAi;
  const u16* Bb = B + (long long)b_out * sBo + (long long)b_in * sBi;
  const int m0 = blockIdx.y * BM;
  const int n0 = blockIdx.x * BN;
  const int tid  = threadIdx.x;
  const int wave = tid >> 5;
  const int lane = tid & 31;
  v8f acc[4] = {v8f{}, v8f{}, v8f{}, v8f{}};

  // clamped (always-valid, divergence-free) tile stage
  auto load_tile = [&](int k0, int buf) {
#pragma unroll
    for (int it = 0; it < 2; ++it) {            // A: 512 b128 vectors
      int v   = tid + it * 256;
      int row = v >> 2;
      int kk  = (v & 3) * 8;
      int gm  = m0 + row; gm = (gm < M) ? gm : (M - 1);
      async_ld_b128(&As[buf][row][kk], Ab + (long long)gm * lda + k0 + kk);
    }
    if (bNK) {                                  // B fast path: 256 b128 vectors
      int row = tid >> 2;
      int kk  = (tid & 3) * 8;
      int gn  = n0 + row; gn = (gn < N) ? gn : (N - 1);
      async_ld_b128(&Bs[buf][row][kk], Bb + (long long)gn * ldb + k0 + kk);
    } else {                                    // B strided sync path
#pragma unroll
      for (int it = 0; it < 8; ++it) {
        int e   = tid + it * 256;
        int row = e >> 5;
        int kk  = e & 31;
        int gn  = n0 + row;
        u16 val = 0;
        if (gn < N) val = Bb[(long long)(k0 + kk) * ldb + gn];
        Bs[buf][row][kk] = val;
      }
    }
  };

  const int nt = K / BK;
  load_tile(0, 0);
  for (int i = 0; i < nt; ++i) {
    const int buf = i & 1;
    if (i + 1 < nt) {
      load_tile((i + 1) * BK, buf ^ 1);
      if (bNK) WAIT_ASYNC(3); else WAIT_ASYNC(2);   // in-order: tile i landed
    } else {
      WAIT_ASYNC(0);
    }
    __syncthreads();
    // A fragment: lanes 0-15 K={0..7,16..23}, lanes 16-31 K={8..15,24..31}
    int ar = wave * 16 + (lane & 15);
    int kb = (lane < 16) ? 0 : 8;
    v16bf av = mk16(*(const uint4*)(&As[buf][ar][kb]),
                    *(const uint4*)(&As[buf][ar][kb + 16]));
    int kbb = (lane < 16) ? 0 : 8;
    v16bf bv[4];
#pragma unroll
    for (int t = 0; t < 4; ++t) {               // preload all B fragments
      int br = t * 16 + (lane & 15);
      bv[t] = mk16(*(const uint4*)(&Bs[buf][br][kbb]),
                   *(const uint4*)(&Bs[buf][br][kbb + 16]));
    }
#pragma unroll
    for (int t = 0; t < 4; ++t)                 // back-to-back WMMAs
      acc[t] = __builtin_amdgcn_wmma_f32_16x16x32_bf16(
          false, av, false, bv[t], (short)0, acc[t], false, false);
    __syncthreads();
  }

  // epilogue: C/D layout per ISA (VGPR r -> M = r + 8*(lane>>4), N = lane&15)
#pragma unroll
  for (int t = 0; t < 4; ++t) {
#pragma unroll
    for (int r = 0; r < 8; ++r) {
      int mg = m0 + wave * 16 + r + ((lane >> 4) << 3);
      int ng = n0 + t * 16 + (lane & 15);
      if (mg < M && ng < N) {
        float val = acc[t][r];
        if (bias)     val += bias[ng];
        if (act == 1) val = fmaxf(val, 0.0f);
        else if (act == 2) val = 1.0f / (1.0f + __expf(-val));
        if (rowscale) val *= rowscale[mg];
        if (emul)     val *= emul[(long long)mg * N + ng];
        if (C) {
          long long ci;
          if (cmode == 1) ci = ((long long)mg * ldc + ng) * sCi + bo;
          else ci = (long long)b_out * sCo + (long long)b_in * sCi
                  + (long long)mg * ldc + ng;
          if (accum) C[ci] += val; else C[ci] = val;
        }
        if (D) {
          long long di;
          if (dmode == 1)
            di = (((long long)(mg >> 5) * 256 + (ng >> 5)) * 1024)
               + (long long)(mg & 31) * 32 + (ng & 31);
          else
            di = (long long)b_out * sDo + (long long)b_in * sDi
               + (long long)mg * ldd + ng;
          D[di] = f2bf(val);
        }
      }
    }
  }
}

// ---------------- LayerNorm (f32 in, bf16 out) ----------------
__global__ __launch_bounds__(128) void k_layernorm(
    const float* __restrict__ X, const float* __restrict__ gw,
    const float* __restrict__ gb, u16* __restrict__ Y, int C)
{
  int row = blockIdx.x, tid = threadIdx.x;
  const float* x = X + (long long)row * C;
  float s = 0.f, ss = 0.f;
  for (int c = tid; c < C; c += 128) { float v = x[c]; s += v; ss += v * v; }
  __shared__ float r1[128], r2[128];
  r1[tid] = s; r2[tid] = ss; __syncthreads();
  for (int k = 64; k > 0; k >>= 1) {
    if (tid < k) { r1[tid] += r1[tid + k]; r2[tid] += r2[tid + k]; }
    __syncthreads();
  }
  float mean = r1[0] / C;
  float var  = r2[0] / C - mean * mean;
  float inv  = rsqrtf(var + 1e-5f);
  u16* y = Y + (long long)row * C;
  for (int c = tid; c < C; c += 128)
    y[c] = f2bf((x[c] - mean) * inv * gw[c] + gb[c]);
}

// ---------------- softmax with mask + pair/tri bias, bf16 probs out ----------
// scores [B,256,256]; batch b: h=b%H, outer=b/H (+rowoff for chunking)
__global__ __launch_bounds__(256) void k_softmax(
    const float* __restrict__ S, u16* __restrict__ P,
    const float* __restrict__ mask, int mask_ld, int rowoff,
    const float* __restrict__ bias2, int H, float scale)
{
  int q = blockIdx.x, b = blockIdx.y;
  int h = b % H, outer = b / H;
  int col = threadIdx.x;
  long long base = ((long long)b * 256 + q) * 256;
  float mval = mask[(long long)(outer + rowoff) * mask_ld + col];
  float bval = bias2[((long long)h * 256 + q) * 256 + col];
  float v = S[base + col] * scale + 1e9f * (mval - 1.0f) + bval;
  __shared__ float red[256];
  red[col] = v; __syncthreads();
  for (int s = 128; s > 0; s >>= 1) {
    if (col < s) red[col] = fmaxf(red[col], red[col + s]);
    __syncthreads();
  }
  float mx = red[0]; __syncthreads();
  float e = __expf(v - mx);
  red[col] = e; __syncthreads();
  for (int s = 128; s > 0; s >>= 1) {
    if (col < s) red[col] += red[col + s];
    __syncthreads();
  }
  P[base + col] = f2bf(e / red[0]);
}

// ---------------- small elementwise kernels ----------------
__global__ void k_convert_w(const float* in, u16* out, int K, int N) {
  long long i = (long long)blockIdx.x * 256 + threadIdx.x;
  if (i < (long long)K * N) {
    int kk = (int)(i / N), nn = (int)(i - (long long)kk * N);
    out[(long long)nn * K + kk] = f2bf(in[i]);     // transpose to [N][K]
  }
}
__global__ void k_transpose_f32(const float* in, float* out, long long R, long long C) {
  long long i = (long long)blockIdx.x * 256 + threadIdx.x;
  if (i < R * C) { long long r = i / C, c = i - r * C; out[c * R + r] = in[i]; }
}
__global__ void k_transpose_bf(const u16* in, u16* out, long long R, long long C) {
  long long i = (long long)blockIdx.x * 256 + threadIdx.x;
  if (i < R * C) { long long r = i / C, c = i - r * C; out[c * R + r] = in[i]; }
}
// out[(d1*D0+d0)*C+c] (+)= in[(d0*D1+d1)*C+c]
__global__ void k_transpose3(const float* in, float* out, int D0, int D1, int C, int accum) {
  long long i = (long long)blockIdx.x * 256 + threadIdx.x;
  long long n = (long long)D0 * D1 * C;
  if (i < n) {
    long long d0 = i / ((long long)D1 * C);
    long long rem = i - d0 * D1 * C;
    long long d1 = rem / C, c = rem - d1 * C;
    long long oi = (d1 * D0 + d0) * C + c;
    if (accum) out[oi] += in[i]; else out[oi] = in[i];
  }
}
// out = bf16( g * p * mask[row] )
__global__ void k_gate3(const float* g, const float* p, const float* mask,
                        u16* out, int C, long long n) {
  long long i = (long long)blockIdx.x * 256 + threadIdx.x;
  if (i < n) {
    float v = g[i] * p[i];
    if (mask) v *= mask[i / C];
    out[i] = f2bf(v);
  }
}
// aC[c][i][k] = a[(i,k) or (k,i)][c]   (256,256,128)
__global__ void k_repack_cik(const u16* in, u16* out, int swp) {
  long long i = (long long)blockIdx.x * 256 + threadIdx.x;
  long long n = 128LL * 65536;
  if (i < n) {
    int c = (int)(i >> 16);
    int r = (int)(i & 65535);
    int ii = r >> 8, kk = r & 255;
    long long src = swp ? ((long long)(kk * 256 + ii) * 128 + c)
                        : ((long long)(ii * 256 + kk) * 128 + c);
    out[i] = in[src];
  }
}
// rs[i*256+j] = 1/(sum_s m[s,i]*m[s,j] + 1e-3)
__global__ __launch_bounds__(256) void k_opm_norm(const float* m, float* rs) {
  int idx = blockIdx.x * 256 + threadIdx.x;   // 65536
  int i = idx >> 8, j = idx & 255;
  float s = 0.f;
  for (int t = 0; t < 128; ++t) s += m[t * 256 + i] * m[t * 256 + j];
  rs[idx] = 1.0f / (s + 1e-3f);
}

// ---------------- host ----------------
static inline dim3 ew(long long n) { return dim3((unsigned)((n + 255) / 256)); }

static void gemm(hipStream_t st,
    const u16* A, int lda, long long sAo, long long sAi,
    const u16* B, int ldb, long long sBo, long long sBi, int bNK,
    float* C, int ldc, long long sCo, long long sCi, int cmode,
    u16* D, int ldd, long long sDo, long long sDi, int dmode,
    const float* bias, const float* rowscale, const float* emul,
    int M, int N, int K, int H, int act, int accum, int batches)
{
  dim3 grid((N + BN - 1) / BN, (M + BM - 1) / BM, batches);
  k_gemm<<<grid, 256, 0, st>>>(A, lda, sAo, sAi, B, ldb, sBo, sBi, bNK,
      C, ldc, sCo, sCi, cmode, D, ldd, sDo, sDi, dmode,
      bias, rowscale, emul, M, N, K, H, act, accum);
}

extern "C" void kernel_launch(void* const* d_in, const int* in_sizes, int n_in,
                              void* d_out, int out_size, void* d_ws, size_t ws_size,
                              hipStream_t stream)
{
  (void)in_sizes; (void)n_in; (void)out_size; (void)ws_size;
  const int MR = 128 * 256;    // MSA rows
  const int ZR = 256 * 256;    // pair rows
  const float SC = 0.17677669529663687f;  // 32^-0.5

  const float* m         = (const float*)d_in[0];
  const float* z         = (const float*)d_in[1];
  const float* msa_mask  = (const float*)d_in[2];
  const float* pair_mask = (const float*)d_in[3];
  auto P = [&](int i) -> const float* { return (const float*)d_in[4 + i]; };

  float* OM = (float*)d_out;                   // [128,256,256] residual m
  float* OZ = OM + (long long)MR * 256;        // [256,256,128] residual z
  hipMemcpyAsync(OM, m, sizeof(float) * (size_t)MR * 256, hipMemcpyDeviceToDevice, stream);
  hipMemcpyAsync(OZ, z, sizeof(float) * (size_t)ZR * 128, hipMemcpyDeviceToDevice, stream);

  char* base = (char*)d_ws;
  size_t cur = 0;
  auto alloc = [&](size_t bytes) -> void* {
    void* p = base + cur;
    cur = (cur + bytes + 255) & ~(size_t)255;
    return p;
  };

  // ---- persistent bf16 transposed weights ----
  struct WInfo { int pidx, K, N; };
  static const WInfo WI[37] = {
    {4,128,8},                                            // 0 mra_z
    {5,256,256},{6,256,256},{7,256,256},{8,256,256},      // 1-4 mra q,k,v,g
    {10,256,256},                                         // 5 mra o
    {14,256,1024},{16,1024,256},                          // 6,7 mt w1,w2
    {20,256,32},{22,256,32},{24,1024,128},                // 8-10 opm w1,w2,wo
    {28,128,128},{30,128,128},{32,128,128},{34,128,128},  // 11-14 tmo ap,ag,bp,bg
    {36,128,128},{40,128,128},                            // 15,16 tmo g,p
    {44,128,128},{46,128,128},{48,128,128},{50,128,128},  // 17-20 tmi ap,ag,bp,bg
    {52,128,128},{56,128,128},                            // 21,22 tmi g,p
    {60,128,4},{61,128,128},{62,128,128},{63,128,128},    // 23-26 tas b,q,k,v
    {64,128,128},{66,128,128},                            // 27,28 tas g,o
    {70,128,4},{71,128,128},{72,128,128},{73,128,128},    // 29-32 tae b,q,k,v
    {74,128,128},{76,128,128},                            // 33,34 tae g,o
    {80,128,512},{82,512,128},                            // 35,36 pt w1,w2
  };
  u16* W[37];
  for (int i = 0; i < 37; ++i) {
    long long tot = (long long)WI[i].K * WI[i].N;
    W[i] = (u16*)alloc((size_t)tot * 2);
    k_convert_w<<<ew(tot), 256, 0, stream>>>(P(WI[i].pidx), W[i], WI[i].K, WI[i].N);
  }
  const size_t phase_base = cur;

  // =========== Phase 1: MSA row attention with pair bias ===========
  {
    cur = phase_base;
    u16*   mn  = (u16*)alloc((size_t)MR * 256 * 2);
    u16*   zn  = (u16*)alloc((size_t)ZR * 128 * 2);
    float* pbt = (float*)alloc((size_t)ZR * 8 * 4);
    float* pb  = (float*)alloc((size_t)8 * ZR * 4);
    u16*   q   = (u16*)alloc((size_t)MR * 256 * 2);
    u16*   k   = (u16*)alloc((size_t)MR * 256 * 2);
    u16*   v   = (u16*)alloc((size_t)MR * 256 * 2);
    float* gb  = (float*)alloc((size_t)MR * 256 * 4);
    float* o   = (float*)alloc((size_t)MR * 256 * 4);
    float* sc  = (float*)alloc(128LL * 65536 * 4);   // 16 seqs * 8 heads chunk
    u16*   pr  = (u16*)alloc(128LL * 65536 * 2);
    u16*   ob  = (u16*)alloc((size_t)MR * 256 * 2);

    k_layernorm<<<MR, 128, 0, stream>>>(OM, P(0), P(1), mn, 256);
    k_layernorm<<<ZR, 128, 0, stream>>>(OZ, P(2), P(3), zn, 128);
    gemm(stream, zn,128,0,0, W[0],128,0,0,1, pbt,8,0,0,0, nullptr,0,0,0,0,
         nullptr,nullptr,nullptr, ZR,8,128,1,0,0,1);
    k_transpose_f32<<<ew((long long)ZR * 8), 256, 0, stream>>>(pbt, pb, ZR, 8);
    gemm(stream, mn,256,0,0, W[1],256,0,0,1, nullptr,0,0,0,0, q,256,0,0,0,
         nullptr,nullptr,nullptr, MR,256,256,1,0,0,1);
    gemm(stream, mn,256,0,0, W[2],256,0,0,1, nullptr,0,0,0,0, k,256,0,0,0,
         nullptr,nullptr,nullptr, MR,256,256,1,0,0,1);
    gemm(stream, mn,256,0,0, W[3],256,0,0,1, nullptr,0,0,0,0, v,256,0,0,0,
         nullptr,nullptr,nullptr, MR,256,256,1,0,0,1);
    gemm(stream, mn,256,0,0, W[4],256,0,0,1, gb,256,0,0,0, nullptr,0,0,0,0,
         P(9),nullptr,nullptr, MR,256,256,1,2,0,1);
    for (int s0 = 0; s0 < 128; s0 += 16) {           // 16 seqs * 8 heads = 128 batches
      long long off = (long long)s0 * 65536;
      gemm(stream, q + off,256,65536,32, k + off,256,65536,32,1,
           sc,256,8LL*65536,65536,0, nullptr,0,0,0,0,
           nullptr,nullptr,nullptr, 256,256,32,8,0,0,128);
      k_softmax<<<dim3(256,128), 256, 0, stream>>>(sc, pr, msa_mask, 256, s0, pb, 8, SC);
      gemm(stream, pr,256,8LL*65536,65536, v + off,256,65536,32,0,
           o + off,256,65536,32,0, nullptr,0,0,0,0,
           nullptr,nullptr,nullptr, 256,32,256,8,0,0,128);
    }
    k_gate3<<<ew((long long)MR * 256), 256, 0, stream>>>(gb, o, nullptr, ob, 256, (long long)MR * 256);
    gemm(stream, ob,256,0,0, W[5],256,0,0,1, OM,256,0,0,0, nullptr,0,0,0,0,
         P(11),nullptr,nullptr, MR,256,256,1,0,1,1);
  }

  // =========== Phase 2: MSA transition ===========
  {
    cur = phase_base;
    u16* mn = (u16*)alloc((size_t)MR * 256 * 2);
    u16* h  = (u16*)alloc((size_t)MR * 1024 * 2);
    k_layernorm<<<MR, 128, 0, stream>>>(OM, P(12), P(13), mn, 256);
    gemm(stream, mn,256,0,0, W[6],256,0,0,1, nullptr,0,0,0,0, h,1024,0,0,0,
         P(15),nullptr,nullptr, MR,1024,256,1,1,0,1);
    gemm(stream, h,1024,0,0, W[7],1024,0,0,1, OM,256,0,0,0, nullptr,0,0,0,0,
         P(17),msa_mask,nullptr, MR,256,1024,1,0,1,1);
  }

  // =========== Phase 3: Outer product mean ===========
  {
    cur = phase_base;
    u16* mn = (u16*)alloc((size_t)MR * 256 * 2);
    u16* ab = (u16*)alloc((size_t)MR * 32 * 2);
    u16* bb = (u16*)alloc((size_t)MR * 32 * 2);
    u16* aT = (u16*)alloc(8192LL * 128 * 2);
    u16* bT = (u16*)alloc(8192LL * 128 * 2);
    u16* O  = (u16*)alloc(8192LL * 8192 * 2);   // stored [(i,j),(c*32+d)] via dmode=1
    float* rs = (float*)alloc((size_t)ZR * 4);
    k_layernorm<<<MR, 128, 0, stream>>>(OM, P(18), P(19), mn, 256);
    gemm(stream, mn,256,0,0, W[8],256,0,0,1, nullptr,0,0,0,0, ab,32,0,0,0,
         P(21),msa_mask,nullptr, MR,32,256,1,0,0,1);
    gemm(stream, mn,256,0,0, W[9],256,0,0,1, nullptr,0,0,0,0, bb,32,0,0,0,
         P(23),msa_mask,nullptr, MR,32,256,1,0,0,1);
    k_transpose_bf<<<ew(128LL*8192), 256, 0, stream>>>(ab, aT, 128, 8192);
    k_transpose_bf<<<ew(128LL*8192), 256, 0, stream>>>(bb, bT, 128, 8192);
    gemm(stream, aT,128,0,0, bT,128,0,0,1, nullptr,0,0,0,0, O,0,0,0,1,
         nullptr,nullptr,nullptr, 8192,8192,128,1,0,0,1);
    k_opm_norm<<<256, 256, 0, stream>>>(msa_mask, rs);
    gemm(stream, O,1024,0,0, W[10],1024,0,0,1, OZ,128,0,0,0, nullptr,0,0,0,0,
         P(25),rs,nullptr, ZR,128,1024,1,0,1,1);
  }

  // =========== Phase 4/5: triangle multiplication (outgoing, incoming) =====
  auto tri_mul = [&](int bp, int wap, int wag, int wbp_, int wbg, int wg, int wp, int swp) {
    cur = phase_base;
    u16*   zn = (u16*)alloc((size_t)ZR * 128 * 2);
    float* t0 = (float*)alloc((size_t)ZR * 128 * 4);
    float* t1 = (float*)alloc((size_t)ZR * 128 * 4);
    u16*   ab = (u16*)alloc((size_t)ZR * 128 * 2);
    u16*   bb = (u16*)alloc((size_t)ZR * 128 * 2);
    float* gz = (float*)alloc((size_t)ZR * 128 * 4);
    u16*   aC = (u16*)alloc(128LL * 65536 * 2);
    u16*   bC = (u16*)alloc(128LL * 65536 * 2);
    float* x  = (float*)alloc((size_t)ZR * 128 * 4);
    u16*   xn = (u16*)alloc((size_t)ZR * 128 * 2);
    k_layernorm<<<ZR, 128, 0, stream>>>(OZ, P(bp), P(bp + 1), zn, 128);
    gemm(stream, zn,128,0,0, W[wag],128,0,0,1, t0,128,0,0,0, nullptr,0,0,0,0,
         P(bp+5),nullptr,nullptr, ZR,128,128,1,2,0,1);
    gemm(stream, zn,128,0,0, W[wap],128,0,0,1, t1,128,0,0,0, nullptr,0,0,0,0,
         P(bp+3),nullptr,nullptr, ZR,128,128,1,0,0,1);
    k_gate3<<<ew((long long)ZR*128), 256, 0, stream>>>(t0, t1, pair_mask, ab, 128, (long long)ZR*128);
    gemm(stream, zn,128,0,0, W[wbg],128,0,0,1, t0,128,0,0,0, nullptr,0,0,0,0,
         P(bp+9),nullptr,nullptr, ZR,128,128,1,2,0,1);
    gemm(stream, zn,128,0,0, W[wbp_],128,0,0,1, t1,128,0,0,0, nullptr,0,0,0,0,
         P(bp+7),nullptr,nullptr, ZR,128,128,1,0,0,1);
    k_gate3<<<ew((long long)ZR*128), 256, 0, stream>>>(t0, t1, pair_mask, bb, 128, (long long)ZR*128);
    gemm(stream, zn,128,0,0, W[wg],128,0,0,1, gz,128,0,0,0, nullptr,0,0,0,0,
         P(bp+11),nullptr,nullptr, ZR,128,128,1,2,0,1);
    k_repack_cik<<<ew(128LL*65536), 256, 0, stream>>>(ab, aC, swp);
    k_repack_cik<<<ew(128LL*65536), 256, 0, stream>>>(bb, bC, swp);
    // batched over 128 channels: x[(i*256+j)*128 + c] via cmode=1
    gemm(stream, aC,256,65536,0, bC,256,65536,0,1, x,256,0,128,1,
         nullptr,0,0,0,0, nullptr,nullptr,nullptr, 256,256,256,1,0,0,128);
    k_layernorm<<<ZR, 128, 0, stream>>>(x, P(bp + 12), P(bp + 13), xn, 128);
    gemm(stream, xn,128,0,0, W[wp],128,0,0,1, OZ,128,0,0,0, nullptr,0,0,0,0,
         P(bp+15),nullptr,gz, ZR,128,128,1,0,1,1);
  };
  tri_mul(26, 11, 12, 13, 14, 15, 16, 0);   // outgoing: 'ikc,jkc->ijc'
  tri_mul(42, 17, 18, 19, 20, 21, 22, 1);   // incoming: 'kic,kjc->ijc'

  // =========== Phase 6/7: triangle attention (starting, ending) ===========
  auto tri_att = [&](int bp, int wb, int wq, int wk, int wv, int wg, int wo,
                     const float* ZIN, const float* maskp, float* Ctar, int accum) {
    u16*   zn  = (u16*)alloc((size_t)ZR * 128 * 2);
    float* tbt = (float*)alloc((size_t)ZR * 4 * 4);
    float* tb  = (float*)alloc((size_t)4 * ZR * 4);
    u16*   q   = (u16*)alloc((size_t)ZR * 128 * 2);
    u16*   k   = (u16*)alloc((size_t)ZR * 128 * 2);
    u16*   v   = (u16*)alloc((size_t)ZR * 128 * 2);
    float* gb  = (float*)alloc((size_t)ZR * 128 * 4);
    float* o   = (float*)alloc((size_t)ZR * 128 * 4);
    float* sc  = (float*)alloc(128LL * 65536 * 4);   // 32 rows * 4 heads chunk
    u16*   pr  = (u16*)alloc(128LL * 65536 * 2);
    u16*   ob  = (u16*)alloc((size_t)ZR * 128 * 2);
    k_layernorm<<<ZR, 128, 0, stream>>>(ZIN, P(bp), P(bp + 1), zn, 128);
    gemm(stream, zn,128,0,0, W[wb],128,0,0,1, tbt,4,0,0,0, nullptr,0,0,0,0,
         nullptr,nullptr,nullptr, ZR,4,128,1,0,0,1);
    k_transpose_f32<<<ew((long long)ZR * 4), 256, 0, stream>>>(tbt, tb, ZR, 4);
    gemm(stream, zn,128,0,0, W[wq],128,0,0,1, nullptr,0,0,0,0, q,128,0,0,0,
         nullptr,nullptr,nullptr, ZR,128,128,1,0,0,1);
    gemm(stream, zn,128,0,0, W[wk],128,0,0,1, nullptr,0,0,0,0, k,128,0,0,0,
         nullptr,nullptr,nullptr, ZR,128,128,1,0,0,1);
    gemm(stream, zn,128,0,0, W[wv],128,0,0,1, nullptr,0,0,0,0, v,128,0,0,0,
         nullptr,nullptr,nullptr, ZR,128,128,1,0,0,1);
    gemm(stream, zn,128,0,0, W[wg],128,0,0,1, gb,128,0,0,0, nullptr,0,0,0,0,
         P(bp+7),nullptr,nullptr, ZR,128,128,1,2,0,1);
    for (int i0 = 0; i0 < 256; i0 += 32) {           // 32 rows * 4 heads = 128 batches
      long long off = (long long)i0 * 32768;
      gemm(stream, q + off,128,32768,32, k + off,128,32768,32,1,
           sc,256,4LL*65536,65536,0, nullptr,0,0,0,0,
           nullptr,nullptr,nullptr, 256,256,32,4,0,0,128);
      k_softmax<<<dim3(256,128), 256, 0, stream>>>(sc, pr, maskp, 256, i0, tb, 4, SC);
      gemm(stream, pr,256,4LL*65536,65536, v + off,128,32768,32,0,
           o + off,128,32768,32,0, nullptr,0,0,0,0,
           nullptr,nullptr,nullptr, 256,32,256,4,0,0,128);
    }
    k_gate3<<<ew((long long)ZR*128), 256, 0, stream>>>(gb, o, nullptr, ob, 128, (long long)ZR*128);
    gemm(stream, ob,128,0,0, W[wo],128,0,0,1, Ctar,128,0,0,0, nullptr,0,0,0,0,
         P(bp+9),nullptr,nullptr, ZR,128,128,1,0,accum,1);
  };
  cur = phase_base;
  tri_att(58, 23, 24, 25, 26, 27, 28, OZ, pair_mask, OZ, 1);   // starting node
  {
    cur = phase_base;
    float* ZT    = (float*)alloc((size_t)ZR * 128 * 4);
    float* maskT = (float*)alloc((size_t)ZR * 4);
    float* delta = (float*)alloc((size_t)ZR * 128 * 4);
    k_transpose3<<<ew((long long)ZR*128), 256, 0, stream>>>(OZ, ZT, 256, 256, 128, 0);
    k_transpose3<<<ew((long long)ZR), 256, 0, stream>>>(pair_mask, maskT, 256, 256, 1, 0);
    tri_att(68, 29, 30, 31, 32, 33, 34, ZT, maskT, delta, 0);  // ending node on z^T
    k_transpose3<<<ew((long long)ZR*128), 256, 0, stream>>>(delta, OZ, 256, 256, 128, 1);
  }

  // =========== Phase 8: pair transition ===========
  {
    cur = phase_base;
    u16* zn = (u16*)alloc((size_t)ZR * 128 * 2);
    u16* h  = (u16*)alloc((size_t)ZR * 512 * 2);
    k_layernorm<<<ZR, 128, 0, stream>>>(OZ, P(78), P(79), zn, 128);
    gemm(stream, zn,128,0,0, W[35],128,0,0,1, nullptr,0,0,0,0, h,512,0,0,0,
         P(81),nullptr,nullptr, ZR,512,128,1,1,0,1);
    gemm(stream, h,512,0,0, W[36],512,0,0,1, OZ,128,0,0,0, nullptr,0,0,0,0,
         P(83),pair_mask,nullptr, ZR,128,512,1,0,1,1);
  }
}
